// DeepKNNModel_26774826123303
// MI455X (gfx1250) — matrix operand order repaired and checked by reading, compiled, not run on hardware
//
#include <hip/hip_runtime.h>
#include <hip/hip_bf16.h>

typedef __attribute__((ext_vector_type(16))) _Float16 v16h;
typedef __attribute__((ext_vector_type(8)))  _Float16 v8h;
typedef __attribute__((ext_vector_type(8)))  float    v8f;
typedef __attribute__((ext_vector_type(4)))  unsigned v4u;
typedef __attribute__((ext_vector_type(8)))  unsigned v8u;

#define B_  512
#define C_  64
#define F_  32
#define NF_ 48
#define PE_ 64
#define D_  512
#define DB_ 1024
#define NC_ 10
#define NTOT 33280   // B + B*C rows through encode
#define NCTX 32768   // B*C

union V16U { v16h v; v8h h[2]; };

__device__ __forceinline__ unsigned lds_off(const void* p) {
    return (unsigned)(unsigned long long)p;   // low 32 bits of generic addr = LDS offset
}

// ---------------------------------------------------------------------------
// f32 -> f16 weight conversion
// ---------------------------------------------------------------------------
__global__ void conv_f32_f16(const float* __restrict__ src, _Float16* __restrict__ dst, int n) {
    int i = blockIdx.x * 256 + threadIdx.x;
    if (i < n) dst[i] = (_Float16)src[i];
}

// ---------------------------------------------------------------------------
// Pipelined WMMA GEMM:  out[M,N] = act( A[M,K] * W[N,K]^T + bias ) (+res)
// Block: 256 threads (8 waves). Tile: 128(M) x 128(N), K-step 32.
// Double-buffered LDS; tile k+1 is issued (TDM for W, async-to-LDS for A)
// before tile k's counters are drained: wait asynccnt<=2 / tensorcnt<=1
// exploits in-order completion per counter to overlap DMA with WMMA.
// Wave tile 32x64 -> 8 wmma per wave per K-step.
// Requires: M % 128 == 0, N % 128 == 0, K % 32 == 0 (true for all call sites).
// ---------------------------------------------------------------------------
template<bool RELU, bool BIAS, bool RES, bool OUT32, bool OUT16>
__global__ void gemm_wmma(const _Float16* __restrict__ A, const _Float16* __restrict__ W,
                          const float* __restrict__ bias, const float* __restrict__ res,
                          float* __restrict__ out32, _Float16* __restrict__ out16,
                          int M, int N, int K) {
    __shared__ _Float16 As[2][128][48];  // 48-half stride: rows 96B, 16B-aligned frags
    __shared__ _Float16 Bs[2][128][32];  // dense 128x32 tiles, filled by the TDM
    const int tid  = threadIdx.x;
    const int m0   = blockIdx.x * 128;
    const int n0   = blockIdx.y * 128;
    const int wave = tid >> 5, lane = tid & 31;
    const int wm = wave >> 1, wn = wave & 1;

    v8f acc[2][4] = {};

    const int arow = tid >> 1, acol = (tid & 1) * 16;   // 128 rows x 32 halves
    const unsigned bsBase = lds_off(&Bs[0][0][0]);

    // issue TDM (W tile) + async loads (A tile) for K-chunk kt into buffer buf
    auto issue_tile = [&](int kt, int buf) {
        if (wave == 0) {
            unsigned long long gaddr = (unsigned long long)(W + (size_t)n0 * K + kt * 32);
            v4u g0 = { 1u,                                    // count=1 (valid D#)
                       bsBase + (unsigned)buf * 8192u,        // lds_addr
                       (unsigned)gaddr,                       // global_addr[31:0]
                       (unsigned)(gaddr >> 32) | 0x80000000u  // global_addr hi | type=2
                     };
            unsigned uK = (unsigned)K;
            unsigned uN = (unsigned)N;
            v8u g1 = { 0x00010000u,                           // wg_mask=0, data_size=2B
                       (uK & 0xFFFFu) << 16,                  // tensor_dim0 lo16
                       (uK >> 16) | ((uN & 0xFFFFu) << 16),   // tensor_dim0 hi | tensor_dim1 lo
                       (uN >> 16) | (32u << 16),              // tensor_dim1 hi | tile_dim0=32
                       128u,                                  // tile_dim1=128, tile_dim2=0
                       uK,                                    // tensor_dim0_stride lo32
                       0u, 0u };
            v4u g2 = { 0u, 0u, 0u, 0u };
            v4u g3 = { 0u, 0u, 0u, 0u };
            asm volatile("tensor_load_to_lds %0, %1, %2, %3"
                         :: "s"(g0), "s"(g1), "s"(g2), "s"(g3) : "memory");
        }
        const _Float16* ag = A + (size_t)(m0 + arow) * K + kt * 32 + acol;
        unsigned la = lds_off(&As[buf][arow][acol]);
        asm volatile("global_load_async_to_lds_b128 %0, %1, off"
                     :: "v"(la), "v"(ag) : "memory");
        asm volatile("global_load_async_to_lds_b128 %0, %1, off"
                     :: "v"(la + 16u), "v"(ag + 8) : "memory");
    };

    const int T = K >> 5;
    issue_tile(0, 0);

    for (int kt = 0; kt < T; kt++) {
        const int buf = kt & 1;
        if (kt + 1 < T) {
            issue_tile(kt + 1, buf ^ 1);            // overlap next DMA with this compute
            asm volatile("s_wait_asynccnt 0x2" ::: "memory");   // tile kt resident
            if (wave == 0) __builtin_amdgcn_s_wait_tensorcnt(1);
        } else {
            asm volatile("s_wait_asynccnt 0x0" ::: "memory");
            if (wave == 0) __builtin_amdgcn_s_wait_tensorcnt(0);
        }
        __syncthreads();

        V16U a[2], b[4];
        #pragma unroll
        for (int ms = 0; ms < 2; ms++) {
            int row = wm * 32 + ms * 16 + (lane & 15);
            int kb  = (lane < 16) ? 0 : 8;               // ISA 16-bit A layout
            a[ms].h[0] = *(const v8h*)&As[buf][row][kb];
            a[ms].h[1] = *(const v8h*)&As[buf][row][kb + 16];
        }
        #pragma unroll
        for (int ns = 0; ns < 4; ns++) {
            int col = wn * 64 + ns * 16 + (lane & 15);
            int kb  = (lane < 16) ? 0 : 8;               // ISA 16-bit B layout
            b[ns].h[0] = *(const v8h*)&Bs[buf][col][kb];
            b[ns].h[1] = *(const v8h*)&Bs[buf][col][kb + 16];
        }
        #pragma unroll
        for (int ms = 0; ms < 2; ms++)
            #pragma unroll
            for (int ns = 0; ns < 4; ns++)
                acc[ms][ns] = __builtin_amdgcn_wmma_f32_16x16x32_f16(
                    false, a[ms].v, false, b[ns].v, (short)0, acc[ms][ns], false, false);
        __syncthreads();
    }

    #pragma unroll
    for (int ms = 0; ms < 2; ms++)
        #pragma unroll
        for (int ns = 0; ns < 4; ns++) {
            int ncol = n0 + wn * 64 + ns * 16 + (lane & 15);
            float bv = BIAS ? bias[ncol] : 0.0f;
            #pragma unroll
            for (int v = 0; v < 8; v++) {
                int m = m0 + wm * 32 + ms * 16 + v + ((lane >= 16) ? 8 : 0);
                float val = acc[ms][ns][v] + bv;
                if (RELU) val = fmaxf(val, 0.0f);
                size_t oi = (size_t)m * N + ncol;
                if (RES)   val += res[oi];
                if (OUT32) out32[oi] = val;
                if (OUT16) out16[oi] = (_Float16)val;
            }
        }
}

// ---------------------------------------------------------------------------
// PLR embedding: rows = (n,f) pairs (M = NTOT*F_), cos/sin features (K=96)
// WMMA against plr_w [PE,96]; relu(+bias) -> h16[M,64] (== h[N, F*PE])
// ---------------------------------------------------------------------------
__global__ void plr_kernel(const float* __restrict__ x_num, const float* __restrict__ ctx_num,
                           const float* __restrict__ freq, const _Float16* __restrict__ plrw16,
                           const float* __restrict__ plr_b, _Float16* __restrict__ h16) {
    __shared__ _Float16 As[128][112];   // 96 + pad (224B rows, 16B aligned frags)
    __shared__ _Float16 Bs[64][112];
    const int tid = threadIdx.x;
    const int m0  = blockIdx.x * 128;

    for (int idx = tid; idx < 64 * 96; idx += 256)
        Bs[idx / 96][idx % 96] = plrw16[idx];

    for (int idx = tid; idx < 128 * 96; idx += 256) {
        int i = idx / 96, kk = idx % 96;
        int r = m0 + i;
        int n = r >> 5, f = r & 31;
        float xv = (n < B_) ? x_num[(n << 5) + f] : ctx_num[((n - B_) << 5) + f];
        float ang = 6.28318530717958647f * freq[f * NF_ + (kk % NF_)] * xv;
        As[i][kk] = (_Float16)((kk < NF_) ? cosf(ang) : sinf(ang));
    }
    __syncthreads();

    const int wave = tid >> 5, lane = tid & 31;
    const int wm = wave >> 1, wn = wave & 1;
    v8f acc[2][2] = {};

    #pragma unroll
    for (int kt = 0; kt < 3; kt++) {
        int kb = kt * 32;
        V16U a[2], b[2];
        #pragma unroll
        for (int ms = 0; ms < 2; ms++) {
            int row = wm * 32 + ms * 16 + (lane & 15);
            int k2  = kb + ((lane < 16) ? 0 : 8);
            a[ms].h[0] = *(const v8h*)&As[row][k2];
            a[ms].h[1] = *(const v8h*)&As[row][k2 + 16];
        }
        #pragma unroll
        for (int ns = 0; ns < 2; ns++) {
            int col = wn * 32 + ns * 16 + (lane & 15);
            int k2  = kb + ((lane < 16) ? 0 : 8);
            b[ns].h[0] = *(const v8h*)&Bs[col][k2];
            b[ns].h[1] = *(const v8h*)&Bs[col][k2 + 16];
        }
        #pragma unroll
        for (int ms = 0; ms < 2; ms++)
            #pragma unroll
            for (int ns = 0; ns < 2; ns++)
                acc[ms][ns] = __builtin_amdgcn_wmma_f32_16x16x32_f16(
                    false, a[ms].v, false, b[ns].v, (short)0, acc[ms][ns], false, false);
    }

    #pragma unroll
    for (int ms = 0; ms < 2; ms++)
        #pragma unroll
        for (int ns = 0; ns < 2; ns++) {
            int p = wn * 32 + ns * 16 + (lane & 15);
            float bv = plr_b[p];
            #pragma unroll
            for (int v = 0; v < 8; v++) {
                int m = wm * 32 + ms * 16 + v + ((lane >= 16) ? 8 : 0);
                float val = fmaxf(acc[ms][ns][v] + bv, 0.0f);
                h16[(size_t)(m0 + m) * PE_ + p] = (_Float16)val;
            }
        }
}

// ---------------------------------------------------------------------------
// Row LayerNorm (D=512) -> f16 output. One wave per row, 8 rows per block.
// ---------------------------------------------------------------------------
__global__ void ln_kernel(const float* __restrict__ x, const float* __restrict__ g,
                          const float* __restrict__ b, _Float16* __restrict__ out16, int M) {
    const int wave = threadIdx.x >> 5, lane = threadIdx.x & 31;
    const int row = blockIdx.x * 8 + wave;
    if (row >= M) return;
    const float* xr = x + (size_t)row * D_;
    float vals[16], s = 0.f, ss = 0.f;
    #pragma unroll
    for (int i = 0; i < 16; i++) {
        float v = xr[lane + i * 32];
        vals[i] = v; s += v; ss += v * v;
    }
    #pragma unroll
    for (int m = 16; m > 0; m >>= 1) { s += __shfl_xor(s, m, 32); ss += __shfl_xor(ss, m, 32); }
    float mean = s * (1.0f / D_);
    float var  = ss * (1.0f / D_) - mean * mean;
    float rs   = rsqrtf(var + 1e-5f);
    #pragma unroll
    for (int i = 0; i < 16; i++) {
        int d = lane + i * 32;
        out16[(size_t)row * D_ + d] = (_Float16)((vals[i] - mean) * rs * g[d] + b[d]);
    }
}

// ---------------------------------------------------------------------------
// sims (neg sq euclid, constant -|k|^2 dropped: softmax-invariant) + softmax
// ---------------------------------------------------------------------------
__global__ void sims_softmax_kernel(const float* __restrict__ kbuf, float* __restrict__ probs) {
    __shared__ float sims[C_];
    const int b = blockIdx.x;
    const int wave = threadIdx.x >> 5, lane = threadIdx.x & 31;
    const float* kq = kbuf + (size_t)b * D_;
    for (int c = wave; c < C_; c += 8) {
        const float* kc = kbuf + (size_t)(B_ + b * C_ + c) * D_;
        float s1 = 0.f, s2 = 0.f;
        #pragma unroll
        for (int i = 0; i < 16; i++) {
            int d = lane + i * 32;
            float q = kq[d], cc = kc[d];
            s1 += q * cc; s2 += cc * cc;
        }
        #pragma unroll
        for (int m = 16; m > 0; m >>= 1) { s1 += __shfl_xor(s1, m, 32); s2 += __shfl_xor(s2, m, 32); }
        if (lane == 0) sims[c] = 2.0f * s1 - s2;
    }
    __syncthreads();
    if (threadIdx.x < 32) {
        float v0 = sims[lane], v1 = sims[lane + 32];
        float mx = fmaxf(v0, v1);
        #pragma unroll
        for (int m = 16; m > 0; m >>= 1) mx = fmaxf(mx, __shfl_xor(mx, m, 32));
        float e0 = expf(v0 - mx), e1 = expf(v1 - mx);
        float s = e0 + e1;
        #pragma unroll
        for (int m = 16; m > 0; m >>= 1) s += __shfl_xor(s, m, 32);
        float inv = 1.0f / s;
        probs[b * C_ + lane]      = e0 * inv;
        probs[b * C_ + lane + 32] = e1 * inv;
    }
}

// diff16[(b,c),d] = k[b,d] - ctx_k[(b,c),d]
__global__ void diff_kernel(const float* __restrict__ kbuf, _Float16* __restrict__ diff16) {
    size_t i = (size_t)blockIdx.x * 256 + threadIdx.x;
    size_t rc = i >> 9; int d = (int)(i & 511);
    int b = (int)(rc >> 6);
    float q  = kbuf[(size_t)b * D_ + d];
    float cc = kbuf[(B_ + rc) * D_ + d];
    diff16[i] = (_Float16)(q - cc);
}

// ctx_x = sum_c probs * (label_emb[ctx_y] + t2) ; x += ctx_x
__global__ void ctx_combine_kernel(const float* __restrict__ probs, const int* __restrict__ ctx_y,
                                   const float* __restrict__ lemb, const float* __restrict__ t2,
                                   float* __restrict__ x32, _Float16* __restrict__ x16) {
    int i = blockIdx.x * 256 + threadIdx.x;
    int b = i >> 9, d = i & 511;
    float acc = 0.0f;
    for (int c = 0; c < C_; c++) {
        int   rc = b * C_ + c;
        float p  = probs[rc];
        int   yi = ctx_y[rc];
        acc += p * (lemb[(size_t)yi * D_ + d] + t2[(size_t)rc * D_ + d]);
    }
    size_t oi = (size_t)b * D_ + d;
    float v = x32[oi] + acc;
    x32[oi] = v;
    x16[oi] = (_Float16)v;
}

// head: logits = relu(LN(x)) @ head_w^T + head_bias
__global__ void head_kernel(const float* __restrict__ x, const float* __restrict__ g,
                            const float* __restrict__ bt, const float* __restrict__ hw,
                            const float* __restrict__ hb, float* __restrict__ out) {
    __shared__ float row[D_];
    __shared__ float sred[8], ssred[8];
    const int b = blockIdx.x, tid = threadIdx.x;
    const int wave = tid >> 5, lane = tid & 31;
    float v0 = x[(size_t)b * D_ + tid];
    float v1 = x[(size_t)b * D_ + 256 + tid];
    float s = v0 + v1, ss = v0 * v0 + v1 * v1;
    #pragma unroll
    for (int m = 16; m > 0; m >>= 1) { s += __shfl_xor(s, m, 32); ss += __shfl_xor(ss, m, 32); }
    if (lane == 0) { sred[wave] = s; ssred[wave] = ss; }
    __syncthreads();
    if (tid == 0) {
        float S = 0.f, SS = 0.f;
        for (int i = 0; i < 8; i++) { S += sred[i]; SS += ssred[i]; }
        sred[0] = S * (1.0f / D_); ssred[0] = SS * (1.0f / D_);
    }
    __syncthreads();
    float mean = sred[0];
    float rs = rsqrtf(ssred[0] - mean * mean + 1e-5f);
    row[tid]       = fmaxf((v0 - mean) * rs * g[tid] + bt[tid], 0.0f);
    row[tid + 256] = fmaxf((v1 - mean) * rs * g[tid + 256] + bt[tid + 256], 0.0f);
    __syncthreads();
    if (tid < NC_) {
        float acc = hb[tid];
        for (int d = 0; d < D_; d++) acc += row[d] * hw[tid * D_ + d];
        out[b * NC_ + tid] = acc;
    }
}

// ---------------------------------------------------------------------------
extern "C" void kernel_launch(void* const* d_in, const int* in_sizes, int n_in,
                              void* d_out, int out_size, void* d_ws, size_t ws_size,
                              hipStream_t stream) {
    (void)in_sizes; (void)n_in; (void)out_size; (void)ws_size;
    const float* x_num   = (const float*)d_in[0];
    const float* ctx_num = (const float*)d_in[1];
    const int*   ctx_y   = (const int*)d_in[2];
    const float* freq    = (const float*)d_in[3];
    const float* plr_w   = (const float*)d_in[4];
    const float* plr_b   = (const float*)d_in[5];
    const float* lin_w   = (const float*)d_in[6];
    const float* lin_b   = (const float*)d_in[7];
    const float* b00_w1  = (const float*)d_in[8];
    const float* b00_b1  = (const float*)d_in[9];
    const float* b00_w2  = (const float*)d_in[10];
    const float* b00_b2  = (const float*)d_in[11];
    const float* b01_w1  = (const float*)d_in[12];
    const float* b01_b1  = (const float*)d_in[13];
    const float* b01_w2  = (const float*)d_in[14];
    const float* b01_b2  = (const float*)d_in[15];
    const float* b01_g   = (const float*)d_in[16];
    const float* b01_bt  = (const float*)d_in[17];
    const float* norm_g  = (const float*)d_in[18];
    const float* norm_b  = (const float*)d_in[19];
    const float* K_w     = (const float*)d_in[20];
    const float* K_b     = (const float*)d_in[21];
    const float* lemb    = (const float*)d_in[22];
    const float* T_w1    = (const float*)d_in[23];
    const float* T_b1    = (const float*)d_in[24];
    const float* T_w2    = (const float*)d_in[25];
    const float* b10_w1  = (const float*)d_in[26];
    const float* b10_b1  = (const float*)d_in[27];
    const float* b10_w2  = (const float*)d_in[28];
    const float* b10_b2  = (const float*)d_in[29];
    const float* b10_g   = (const float*)d_in[30];
    const float* b10_bt  = (const float*)d_in[31];
    const float* b11_w1  = (const float*)d_in[32];
    const float* b11_b1  = (const float*)d_in[33];
    const float* b11_w2  = (const float*)d_in[34];
    const float* b11_b2  = (const float*)d_in[35];
    const float* b11_g   = (const float*)d_in[36];
    const float* b11_bt  = (const float*)d_in[37];
    const float* head_g  = (const float*)d_in[38];
    const float* head_b  = (const float*)d_in[39];
    const float* head_w  = (const float*)d_in[40];
    const float* head_bs = (const float*)d_in[41];
    float* out = (float*)d_out;

    // ---- carve workspace ----
    char* ws = (char*)d_ws;
    size_t off = 0;
    auto take = [&](size_t bytes) -> char* {
        char* p = ws + off;
        off += (bytes + 255) & ~(size_t)255;
        return p;
    };
    _Float16* plrw16  = (_Float16*)take((size_t)PE_ * 2 * NF_ * 2);
    _Float16* linw16  = (_Float16*)take((size_t)D_ * (F_ * PE_) * 2);
    _Float16* w00a    = (_Float16*)take((size_t)DB_ * D_ * 2);
    _Float16* w00b    = (_Float16*)take((size_t)D_ * DB_ * 2);
    _Float16* w01a    = (_Float16*)take((size_t)DB_ * D_ * 2);
    _Float16* w01b    = (_Float16*)take((size_t)D_ * DB_ * 2);
    _Float16* kw16    = (_Float16*)take((size_t)D_ * D_ * 2);
    _Float16* tw1_16  = (_Float16*)take((size_t)DB_ * D_ * 2);
    _Float16* tw2_16  = (_Float16*)take((size_t)D_ * DB_ * 2);
    _Float16* w10a    = (_Float16*)take((size_t)DB_ * D_ * 2);
    _Float16* w10b    = (_Float16*)take((size_t)D_ * DB_ * 2);
    _Float16* w11a    = (_Float16*)take((size_t)DB_ * D_ * 2);
    _Float16* w11b    = (_Float16*)take((size_t)D_ * DB_ * 2);
    _Float16* h16     = (_Float16*)take((size_t)NTOT * (F_ * PE_) * 2);
    float*    x32     = (float*)   take((size_t)NTOT * D_ * 4);
    _Float16* x16     = (_Float16*)take((size_t)NTOT * D_ * 2);
    _Float16* ln16    = (_Float16*)take((size_t)NTOT * D_ * 2);
    _Float16* hid16   = (_Float16*)take((size_t)NTOT * DB_ * 2);
    float*    k32     = (float*)   take((size_t)NTOT * D_ * 4);
    _Float16* diff16  = (_Float16*)take((size_t)NCTX * D_ * 2);
    float*    t2_32   = (float*)   take((size_t)NCTX * D_ * 4);
    float*    probs   = (float*)   take((size_t)B_ * C_ * 4);

    // ---- convert weights to f16 ----
    auto conv = [&](const float* s, _Float16* d, int n) {
        conv_f32_f16<<<(n + 255) / 256, 256, 0, stream>>>(s, d, n);
    };
    conv(plr_w, plrw16, PE_ * 2 * NF_);
    conv(lin_w, linw16, D_ * F_ * PE_);
    conv(b00_w1, w00a, DB_ * D_);  conv(b00_w2, w00b, D_ * DB_);
    conv(b01_w1, w01a, DB_ * D_);  conv(b01_w2, w01b, D_ * DB_);
    conv(K_w, kw16, D_ * D_);
    conv(T_w1, tw1_16, DB_ * D_);  conv(T_w2, tw2_16, D_ * DB_);
    conv(b10_w1, w10a, DB_ * D_);  conv(b10_w2, w10b, D_ * DB_);
    conv(b11_w1, w11a, DB_ * D_);  conv(b11_w2, w11b, D_ * DB_);

    const dim3 blk(256);
    // ---- encode (all NTOT rows) ----
    plr_kernel<<<dim3((NTOT * F_) / 128), blk, 0, stream>>>(x_num, ctx_num, freq, plrw16, plr_b, h16);
    gemm_wmma<false, true, false, true, true><<<dim3(NTOT / 128, D_ / 128), blk, 0, stream>>>(
        h16, linw16, lin_b, nullptr, x32, x16, NTOT, D_, F_ * PE_);
    gemm_wmma<true, true, false, false, true><<<dim3(NTOT / 128, DB_ / 128), blk, 0, stream>>>(
        x16, w00a, b00_b1, nullptr, nullptr, hid16, NTOT, DB_, D_);
    gemm_wmma<false, true, true, true, true><<<dim3(NTOT / 128, D_ / 128), blk, 0, stream>>>(
        hid16, w00b, b00_b2, x32, x32, x16, NTOT, D_, DB_);
    ln_kernel<<<dim3(NTOT / 8), blk, 0, stream>>>(x32, b01_g, b01_bt, ln16, NTOT);
    gemm_wmma<true, true, false, false, true><<<dim3(NTOT / 128, DB_ / 128), blk, 0, stream>>>(
        ln16, w01a, b01_b1, nullptr, nullptr, hid16, NTOT, DB_, D_);
    gemm_wmma<false, true, true, true, true><<<dim3(NTOT / 128, D_ / 128), blk, 0, stream>>>(
        hid16, w01b, b01_b2, x32, x32, x16, NTOT, D_, DB_);
    ln_kernel<<<dim3(NTOT / 8), blk, 0, stream>>>(x32, norm_g, norm_b, ln16, NTOT);
    gemm_wmma<false, true, false, true, false><<<dim3(NTOT / 128, D_ / 128), blk, 0, stream>>>(
        ln16, kw16, K_b, nullptr, k32, nullptr, NTOT, D_, D_);

    // ---- kNN attention ----
    sims_softmax_kernel<<<dim3(B_), blk, 0, stream>>>(k32, probs);
    diff_kernel<<<dim3((NCTX * D_) / 256), blk, 0, stream>>>(k32, diff16);
    gemm_wmma<true, true, false, false, true><<<dim3(NCTX / 128, DB_ / 128), blk, 0, stream>>>(
        diff16, tw1_16, T_b1, nullptr, nullptr, hid16, NCTX, DB_, D_);
    gemm_wmma<false, false, false, true, false><<<dim3(NCTX / 128, D_ / 128), blk, 0, stream>>>(
        hid16, tw2_16, nullptr, nullptr, t2_32, nullptr, NCTX, D_, DB_);
    ctx_combine_kernel<<<dim3((B_ * D_) / 256), blk, 0, stream>>>(probs, ctx_y, lemb, t2_32, x32, x16);

    // ---- tail blocks (B rows only) ----
    ln_kernel<<<dim3(B_ / 8), blk, 0, stream>>>(x32, b10_g, b10_bt, ln16, B_);
    gemm_wmma<true, true, false, false, true><<<dim3(B_ / 128, DB_ / 128), blk, 0, stream>>>(
        ln16, w10a, b10_b1, nullptr, nullptr, hid16, B_, DB_, D_);
    gemm_wmma<false, true, true, true, true><<<dim3(B_ / 128, D_ / 128), blk, 0, stream>>>(
        hid16, w10b, b10_b2, x32, x32, x16, B_, D_, DB_);
    ln_kernel<<<dim3(B_ / 8), blk, 0, stream>>>(x32, b11_g, b11_bt, ln16, B_);
    gemm_wmma<true, true, false, false, true><<<dim3(B_ / 128, DB_ / 128), blk, 0, stream>>>(
        ln16, w11a, b11_b1, nullptr, nullptr, hid16, B_, DB_, D_);
    gemm_wmma<false, true, true, true, true><<<dim3(B_ / 128, D_ / 128), blk, 0, stream>>>(
        hid16, w11b, b11_b2, x32, x32, x16, B_, D_, DB_);

    // ---- head ----
    head_kernel<<<dim3(B_), blk, 0, stream>>>(x32, head_g, head_b, head_w, head_bs, out);
}
